// DEP_32779190403558
// MI455X (gfx1250) — matrix operand order repaired and checked by reading, not compiled, not run on hardware
//
#include <hip/hip_runtime.h>
#include <hip/hip_bf16.h>

// ---------------------------------------------------------------------------
// Edge pruning via exact radix-select (replaces full argsort) + WMMA-based
// inter-block prefix scan + bandwidth-bound compaction/gather.
// MI455X reasoning: total traffic ~400MB -> ~17us at 23.3TB/s; the 64MB
// weight matrix is L2-resident (192MB), so the random gathers in pass 1/2
// hit L2. The only "compute" is selection, done with two 65536-bin histogram
// passes. The 1024-entry block-count scan uses v_wmma_f32_16x16x32_f16
// (counts <= 256 are f16-exact; f32 accumulation exact), the CDNA5 path.
// ---------------------------------------------------------------------------

#define INPUT_DIM 4096
#define BATCH     64
#define E_PER     262144
#define NUM_RM    78643                 // int(262144 * 0.3)
#define KEPT      (E_PER - NUM_RM)      // 183501
#define E_TOTAL   (BATCH * E_PER)       // 16777216 (per row of edge_index)
#define NBLK      (E_PER / 256)         // 1024

typedef __attribute__((ext_vector_type(16))) _Float16 v16h;
typedef __attribute__((ext_vector_type(8)))  float    v8f;

__device__ __forceinline__ unsigned sortable_key(float v) {
  unsigned u = __float_as_uint(v);
  return (u & 0x80000000u) ? ~u : (u | 0x80000000u);
}

// ---- Pass 1: per-edge key (monotone-mapped sigmoid) + high-16 histogram ----
__global__ void k_keys(const int* __restrict__ ei, const float* __restrict__ W,
                       unsigned* __restrict__ keys, unsigned* __restrict__ hist) {
  int e = blockIdx.x * blockDim.x + threadIdx.x;
  int s = ei[e];
  int d = ei[E_TOTAL + e];
  float a  = W[s * INPUT_DIM + d];
  float b  = W[d * INPUT_DIM + s];
  float sg = 1.0f / (1.0f + expf(-(a + b)));
  unsigned k = sortable_key(sg);
  keys[e] = k;
  atomicAdd(&hist[k >> 16], 1u);
}

// ---- Radix-select step over a 65536-bin histogram (single block) ----------
// pass 0: find high bucket B1 containing rank NUM_RM; write scal[0]=B1, scal[1]=rem
// pass 1: find low bin for rank scal[1];   write scal[2]=T(full key), scal[3]=r
__global__ void k_select(const unsigned* __restrict__ hist,
                         unsigned* __restrict__ scal, int pass) {
  __shared__ unsigned part[256];
  __shared__ unsigned excl[256];
  int t = threadIdx.x;
  unsigned sum = 0;
  for (int i = 0; i < 256; ++i) sum += hist[t * 256 + i];
  part[t] = sum;
  __syncthreads();
  if (t == 0) {
    unsigned run = 0;
    for (int i = 0; i < 256; ++i) { excl[i] = run; run += part[i]; }
  }
  __syncthreads();
  unsigned target = (pass == 0) ? (unsigned)NUM_RM : scal[1];
  unsigned cb = excl[t];
  if (cb < target && cb + part[t] >= target) {   // exactly one thread
    unsigned run = cb;
    for (int i = 0; i < 256; ++i) {
      unsigned c = hist[t * 256 + i];
      if (run + c >= target) {
        unsigned bin = (unsigned)(t * 256 + i);
        if (pass == 0) { scal[0] = bin; scal[1] = target - run; }
        else           { scal[2] = (scal[0] << 16) | bin; scal[3] = target - run; }
        break;
      }
      run += c;
    }
  }
}

// ---- Pass 2: low-16 histogram restricted to the selected high bucket ------
__global__ void k_hist2(const unsigned* __restrict__ keys,
                        const unsigned* __restrict__ scal,
                        unsigned* __restrict__ hist) {
  int e = blockIdx.x * blockDim.x + threadIdx.x;
  unsigned k = keys[e];
  if ((k >> 16) == scal[0]) atomicAdd(&hist[k & 0xFFFFu], 1u);
}

// ---- Per-block counts of (key < T) and (key == T) -------------------------
__global__ void k_bcounts(const unsigned* __restrict__ keys,
                          const unsigned* __restrict__ scal,
                          unsigned* __restrict__ bcS, unsigned* __restrict__ bcT) {
  __shared__ unsigned cS, cT;
  if (threadIdx.x == 0) { cS = 0; cT = 0; }
  __syncthreads();
  int e = blockIdx.x * 256 + threadIdx.x;
  unsigned T = scal[2];
  unsigned k = keys[e];
  unsigned long long bs = __ballot(k < T);
  unsigned long long bt = __ballot(k == T);
  if ((threadIdx.x & 31) == 0) {
    atomicAdd(&cS, (unsigned)__popcll(bs));
    atomicAdd(&cT, (unsigned)__popcll(bt));
  }
  __syncthreads();
  if (threadIdx.x == 0) { bcS[blockIdx.x] = cS; bcT[blockIdx.x] = cT; }
}

// ---- Exclusive scan of 1024 block counts via v_wmma_f32_16x16x32_f16 ------
// One wave. 256 values per step as V(16x16); D = V x U (upper-tri ones) gives
// row-inclusive scans; row totals broadcast via shfl; f32 carry across groups.
// Counts <= 256 (f16-exact), accumulation f32-exact.
__global__ void k_scan_wmma(const unsigned* __restrict__ bcS,
                            const unsigned* __restrict__ bcT,
                            unsigned* __restrict__ offS, unsigned* __restrict__ offT) {
  int lane = threadIdx.x;
  int hi = lane >> 4;
  int m  = lane & 15;   // A row for this lane
  int n  = lane & 15;   // B column / D column for this lane
  // B = U: U[k][n] = 1 iff k <= n (k < 16). Lane layout: K(h) = h + 16*hi.
  v16h bu;
#pragma unroll
  for (int h = 0; h < 16; ++h) {
    int K = h + 16 * hi;
    bu[h] = (_Float16)((K <= n && K < 16) ? 1.0f : 0.0f);
  }
  for (int which = 0; which < 2; ++which) {
    const unsigned* cnt = which ? bcT : bcS;
    unsigned* out       = which ? offT : offS;
    float carry = 0.0f;
    for (int g = 0; g < 4; ++g) {
      int base = g * 256;
      // A = V: A[m][K] = cnt[base + m*16 + K] for K<16.  K(h) = h + 16*hi.
      v16h a;
#pragma unroll
      for (int h = 0; h < 16; ++h) {
        int K = h + 16 * hi;
        float val = (K < 16) ? (float)cnt[base + m * 16 + K] : 0.0f;
        a[h] = (_Float16)val;
      }
      v8f c = {};
      c = __builtin_amdgcn_wmma_f32_16x16x32_f16(false, a, false, bu,
                                                 (short)0, c, false, false);
      // Row totals D[r][15]: rows 0-7 on lane 15 (c[r]), rows 8-15 on lane 31.
      float rt[16];
#pragma unroll
      for (int j = 0; j < 8; ++j) {
        rt[j]     = __shfl(c[j], 15, 32);
        rt[j + 8] = __shfl(c[j], 31, 32);
      }
      float pre[16];
      pre[0] = 0.0f;
#pragma unroll
      for (int q = 1; q < 16; ++q) pre[q] = pre[q - 1] + rt[q - 1];
      float total = pre[15] + rt[15];
#pragma unroll
      for (int j = 0; j < 8; ++j) {
        int M = j + 8 * hi;                 // D VGPR j -> row M
        int i = base + M * 16 + n;
        float incl = c[j] + pre[M];
        float ex   = carry + incl - (float)cnt[i];   // exclusive
        out[i] = (unsigned)ex;
      }
      carry += total;
    }
  }
}

// ---- Final: mask + stable compaction + 64-batch gather --------------------
__global__ void k_compact(const int* __restrict__ ei, const float* __restrict__ attr,
                          const unsigned* __restrict__ keys,
                          const unsigned* __restrict__ scal,
                          const unsigned* __restrict__ offS,
                          const unsigned* __restrict__ offT,
                          float* __restrict__ out) {
  __shared__ unsigned wS[8], wT[8], wSx[8], wTx[8];
  int t = threadIdx.x;
  int b = blockIdx.x;
  int e = b * 256 + t;
  unsigned T = scal[2];
  unsigned r = scal[3];
  unsigned k = keys[e];
  bool isS = (k < T);
  bool isT = (k == T);
  unsigned long long bS = __ballot(isS);
  unsigned long long bT = __ballot(isT);
  int lane = t & 31, w = t >> 5;
  if (lane == 0) { wS[w] = (unsigned)__popcll(bS); wT[w] = (unsigned)__popcll(bT); }
  __syncthreads();
  if (t == 0) {
    unsigned rs = 0, rt = 0;
    for (int i = 0; i < 8; ++i) { wSx[i] = rs; rs += wS[i]; wTx[i] = rt; rt += wT[i]; }
  }
  __syncthreads();
  unsigned long long lm = (1ull << lane) - 1ull;
  unsigned strictBefore = offS[b] + wSx[w] + (unsigned)__popcll(bS & lm);
  unsigned tieBefore    = offT[b] + wTx[w] + (unsigned)__popcll(bT & lm);
  bool pruned = isS || (isT && tieBefore < r);
  const size_t NK = (size_t)BATCH * KEPT;
  out[3 * NK + (size_t)e] = pruned ? 0.0f : 1.0f;   // mask output
  if (!pruned) {
    unsigned prunedBefore = strictBefore + (tieBefore < r ? tieBefore : r);
    unsigned j = (unsigned)e - prunedBefore;        // stable kept position
    for (int bi = 0; bi < BATCH; ++bi) {
      int off = bi * E_PER + e;
      float si = (float)ei[off];
      float di = (float)ei[E_TOTAL + off];
      float av = attr[off];
      size_t o = (size_t)bi * KEPT + j;
      out[o]          = si;   // out_edge_index row 0
      out[NK + o]     = di;   // out_edge_index row 1
      out[2 * NK + o] = av;   // out_edge_attr
    }
  }
}

extern "C" void kernel_launch(void* const* d_in, const int* in_sizes, int n_in,
                              void* d_out, int out_size, void* d_ws, size_t ws_size,
                              hipStream_t stream) {
  (void)in_sizes; (void)n_in; (void)out_size; (void)ws_size;
  const int*   ei   = (const int*)d_in[0];     // edge_index (2, E) flattened
  const float* attr = (const float*)d_in[1];   // edge_attr  (E,)
  const float* W    = (const float*)d_in[3];   // weight_mask (4096,4096)
  float* out = (float*)d_out;

  unsigned* ws   = (unsigned*)d_ws;
  unsigned* keys = ws;                         // E_PER
  unsigned* hist = keys + E_PER;               // 65536
  unsigned* bcS  = hist + 65536;               // NBLK
  unsigned* bcT  = bcS + NBLK;                 // NBLK
  unsigned* offS = bcT + NBLK;                 // NBLK
  unsigned* offT = offS + NBLK;                // NBLK
  unsigned* scal = offT + NBLK;                // 4 scalars: B1, rem1, T, r

  hipMemsetAsync(hist, 0, 65536 * sizeof(unsigned), stream);
  k_keys   <<<NBLK, 256, 0, stream>>>(ei, W, keys, hist);
  k_select <<<1,    256, 0, stream>>>(hist, scal, 0);
  hipMemsetAsync(hist, 0, 65536 * sizeof(unsigned), stream);
  k_hist2  <<<NBLK, 256, 0, stream>>>(keys, scal, hist);
  k_select <<<1,    256, 0, stream>>>(hist, scal, 1);
  k_bcounts<<<NBLK, 256, 0, stream>>>(keys, scal, bcS, bcT);
  k_scan_wmma<<<1,   32, 0, stream>>>(bcS, bcT, offS, offT);
  k_compact<<<NBLK, 256, 0, stream>>>(ei, attr, keys, scal, offS, offT, out);
}
